// MultiHeadAttention_57569741635879
// MI455X (gfx1250) — matrix operand order repaired and checked by reading, not compiled, not run on hardware
//
#include <hip/hip_runtime.h>
#include <hip/hip_bf16.h>
#include <math.h>

// ---------------------------------------------------------------------------
// MHA forward for MI455X (gfx1250, wave32). All GEMMs via
// v_wmma_f32_16x16x32_bf16 (f32 accumulate). One-time f32->bf16 conversion
// pre-pass so GEMM inner loops are pure b128 loads + WMMA (no per-iteration
// conversion VALU, half the weight bandwidth). Flash-attention streaming
// softmax so the S x S score matrix never touches HBM.
// ---------------------------------------------------------------------------

typedef __attribute__((ext_vector_type(16))) __bf16 v16bf;
typedef __attribute__((ext_vector_type(8)))  float  v8f;

#define D_MODEL 1024
#define N_HEADS 16
#define D_K     64
#define BATCH   4
#define SEQ     2048
#define M_TOTAL (BATCH * SEQ)   // 8192 rows

union Frag16 { v16bf v; unsigned short u[16]; };

__device__ __forceinline__ unsigned int f32_to_bf16_bits(float f) {
  unsigned int u = __float_as_uint(f);
  return (u + 0x7FFFu + ((u >> 16) & 1u)) >> 16;   // round-to-nearest-even
}

__device__ __forceinline__ v8f wmma_bf16(const v16bf& a, const v16bf& b, v8f c) {
#if defined(__gfx1250__)
  return __builtin_amdgcn_wmma_f32_16x16x32_bf16(false, a, false, b, (short)0, c, false, false);
#else
  return c;  // host pass / other arch: never executed
#endif
}

// ---------------------------------------------------------------------------
// Elementwise f32 -> bf16 conversion (vectorized: float4 in, 2x u32 out).
// n must be a multiple of 4 (true for all tensors here).
// ---------------------------------------------------------------------------
__global__ __launch_bounds__(256) void convert_bf16_kernel(
    const float* __restrict__ src, unsigned short* __restrict__ dst, int n4)
{
  int i = blockIdx.x * blockDim.x + threadIdx.x;
  if (i >= n4) return;
  const float4 f = ((const float4*)src)[i];
  uint2 o;
  o.x = f32_to_bf16_bits(f.x) | (f32_to_bf16_bits(f.y) << 16);
  o.y = f32_to_bf16_bits(f.z) | (f32_to_bf16_bits(f.w) << 16);
  ((uint2*)dst)[i] = o;
}

// ---------------------------------------------------------------------------
// C = X @ W^T + bias.  X: [M_TOTAL, 1024] bf16, W: [1024, 1024] bf16
// row-major (nn.Linear layout -> B-fragment lanes read W rows contiguously).
// Each wave: 16x64 tile; block (256 thr = 8 waves): 128x64.
// OUT_MODE 0: bf16 row-major [M,1024]
// OUT_MODE 1: bf16 head-transposed [b, h, d_k, s]   (for V)
// OUT_MODE 2: f32 row-major [M,1024]                (final output)
// ---------------------------------------------------------------------------
template <int OUT_MODE>
__global__ __launch_bounds__(256) void gemm_bias_kernel(
    const unsigned short* __restrict__ X, const unsigned short* __restrict__ W,
    const float* __restrict__ bias, void* __restrict__ outp)
{
  const int lane = threadIdx.x & 31;
  const int wv   = threadIdx.x >> 5;
  const int lg   = lane >> 4;          // 16-lane half-group
  const int ln   = lane & 15;
  const int mrow = blockIdx.x * 128 + wv * 16;
  const int n0   = blockIdx.y * 64;

  v8f acc[4] = {};
  for (int kb = 0; kb < D_MODEL; kb += 32) {
    // A fragment 16x32: lanes0-15 -> K kb..kb+7 / kb+16..kb+23 ; lanes16-31 +8
    Frag16 a;
    const unsigned short* ap = X + (size_t)(mrow + ln) * D_MODEL + kb + lg * 8;
    #pragma unroll
    for (int i = 0; i < 8; ++i) { a.u[i] = ap[i]; a.u[8 + i] = ap[16 + i]; }

    #pragma unroll
    for (int j = 0; j < 4; ++j) {
      // B fragment 32x16: lane = column n, K kb..kb+7 / kb+16..kb+23 (lanes16-31 +8) from W row n
      Frag16 bf;
      const unsigned short* bp = W + (size_t)(n0 + j * 16 + ln) * D_MODEL + kb + lg * 8;
      #pragma unroll
      for (int i = 0; i < 16; ++i) bf.u[i] = bp[(i & 7) + ((i >> 3) << 4)];
      acc[j] = wmma_bf16(a.v, bf.v, acc[j]);
    }
  }

  #pragma unroll
  for (int j = 0; j < 4; ++j) {
    const int   col = n0 + j * 16 + ln;
    const float bn  = bias[col];
    #pragma unroll
    for (int r = 0; r < 8; ++r) {
      const int   m   = mrow + r + lg * 8;     // C layout: lanes16-31 hold M=r+8
      const float val = acc[j][r] + bn;
      if (OUT_MODE == 0) {
        ((unsigned short*)outp)[(size_t)m * D_MODEL + col] =
            (unsigned short)f32_to_bf16_bits(val);
      } else if (OUT_MODE == 1) {
        const int b = m >> 11, s = m & (SEQ - 1);
        const int h = col >> 6, dk = col & (D_K - 1);
        ((unsigned short*)outp)[((size_t)(b * N_HEADS + h) * D_K + dk) * SEQ + s] =
            (unsigned short)f32_to_bf16_bits(val);
      } else {
        ((float*)outp)[(size_t)m * D_MODEL + col] = val;
      }
    }
  }
}

// ---------------------------------------------------------------------------
// Flash attention. grid = (SEQ/128, BATCH*N_HEADS), 256 threads (8 waves).
// Each wave: 16 query rows, streams keys in 64-wide tiles with online softmax.
// Q/K bf16 in [b,s,1024] (head slice), V bf16 head-transposed [b,h,d_k,s].
// ---------------------------------------------------------------------------
__global__ __launch_bounds__(256) void flash_attn_kernel(
    const unsigned short* __restrict__ Qb,
    const unsigned short* __restrict__ Kb,
    const unsigned short* __restrict__ Vt,
    const int* __restrict__ mask,
    unsigned short* __restrict__ Ob)
{
  __shared__ unsigned short ldsP[8 * 16 * 64];   // per-wave 16x64 bf16 P tile
  const int lane = threadIdx.x & 31;
  const int wv   = threadIdx.x >> 5;
  const int lg   = lane >> 4;
  const int ln   = lane & 15;
  const int bh   = blockIdx.y;
  const int b    = bh >> 4, h = bh & 15;
  const int q0   = blockIdx.x * 128 + wv * 16;

  const unsigned short* Qbase = Qb + (size_t)b * SEQ * D_MODEL + h * D_K;
  const unsigned short* Kbase = Kb + (size_t)b * SEQ * D_MODEL + h * D_K;
  const unsigned short* Vbase = Vt + (size_t)(b * N_HEADS + h) * D_K * SEQ;
  const int*            mbase = mask + (size_t)b * SEQ * SEQ;
  unsigned short*       myP   = ldsP + wv * (16 * 64);

  // Q fragments: two 16x32 k-steps over d_k = 64
  Frag16 Aq[2];
  #pragma unroll
  for (int f = 0; f < 2; ++f) {
    const unsigned short* p = Qbase + (size_t)(q0 + ln) * D_MODEL + f * 32 + lg * 8;
    #pragma unroll
    for (int i = 0; i < 8; ++i) { Aq[f].u[i] = p[i]; Aq[f].u[8 + i] = p[16 + i]; }
  }

  float mrow[8], lsum[8];
  #pragma unroll
  for (int r = 0; r < 8; ++r) { mrow[r] = -INFINITY; lsum[r] = 0.0f; }
  v8f accO[4] = {};

  for (int kt = 0; kt < SEQ; kt += 64) {
    // Prefetch next key/value tile into cache (global_prefetch_b8).
    if (kt + 64 < SEQ) {
      __builtin_prefetch(Kbase + (size_t)(kt + 64 + lane) * D_MODEL, 0, 1);
      __builtin_prefetch(Vbase + (size_t)(lane << 1) * SEQ + kt + 64, 0, 1);
    }

    // ---- S = Q @ K^T for 64 keys (4 column tiles x 2 k-steps) ----
    v8f sacc[4] = {};
    #pragma unroll
    for (int j = 0; j < 4; ++j) {
      const int key = kt + j * 16 + ln;          // B lane = key column
      #pragma unroll
      for (int f = 0; f < 2; ++f) {
        Frag16 bk;
        const unsigned short* p = Kbase + (size_t)key * D_MODEL + f * 32 + lg * 8;
        #pragma unroll
        for (int i = 0; i < 16; ++i) bk.u[i] = p[(i & 7) + ((i >> 3) << 4)];
        sacc[j] = wmma_bf16(Aq[f].v, bk.v, sacc[j]);
      }
    }

    // ---- scale, mask, online softmax ----
    float alpha[8];
    #pragma unroll
    for (int r = 0; r < 8; ++r) {
      const int qm = q0 + r + lg * 8;
      float sv[4];
      float tmax = -INFINITY;
      #pragma unroll
      for (int j = 0; j < 4; ++j) {
        float s = sacc[j][r] * 0.125f;           // 1/sqrt(64)
        const int mk = mbase[(size_t)qm * SEQ + kt + j * 16 + ln];
        s = (mk != 0) ? s : -INFINITY;
        sv[j] = s;
        tmax = fmaxf(tmax, s);
      }
      #pragma unroll
      for (int d = 1; d < 16; d <<= 1) tmax = fmaxf(tmax, __shfl_xor(tmax, d, 32));
      const float mnew = fmaxf(mrow[r], tmax);
      const float a    = __expf(mrow[r] - mnew);
      float psum = 0.0f;
      #pragma unroll
      for (int j = 0; j < 4; ++j) {
        const float p = __expf(sv[j] - mnew);
        psum += p;
        myP[(r + lg * 8) * 64 + j * 16 + ln] =
            (unsigned short)f32_to_bf16_bits(p);  // C-layout -> LDS
      }
      #pragma unroll
      for (int d = 1; d < 16; d <<= 1) psum += __shfl_xor(psum, d, 32);
      lsum[r]  = lsum[r] * a + psum;
      mrow[r]  = mnew;
      alpha[r] = a;
    }
    #pragma unroll
    for (int j = 0; j < 4; ++j)
      #pragma unroll
      for (int r = 0; r < 8; ++r) accO[j][r] *= alpha[r];

    // ---- P back out of LDS in A-fragment layout (same wave: LDS in-order) ----
    Frag16 Ap[2];
    #pragma unroll
    for (int f = 0; f < 2; ++f) {
      const unsigned short* p = myP + ln * 64 + f * 32 + lg * 8;
      #pragma unroll
      for (int i = 0; i < 8; ++i) { Ap[f].u[i] = p[i]; Ap[f].u[8 + i] = p[16 + i]; }
    }

    // ---- O += P @ V (V head-transposed: lane = feature, contiguous keys) ----
    #pragma unroll
    for (int j = 0; j < 4; ++j) {
      #pragma unroll
      for (int f = 0; f < 2; ++f) {
        Frag16 bv;
        const unsigned short* p = Vbase + (size_t)(j * 16 + ln) * SEQ + kt + f * 32 + lg * 8;
        #pragma unroll
        for (int i = 0; i < 16; ++i) bv.u[i] = p[(i & 7) + ((i >> 3) << 4)];
        accO[j] = wmma_bf16(Ap[f].v, bv.v, accO[j]);
      }
    }
  }

  // ---- normalize, store bf16 attention output [b, s, 1024] ----
  #pragma unroll
  for (int r = 0; r < 8; ++r) {
    const int   qm  = q0 + r + lg * 8;
    const float inv = 1.0f / lsum[r];
    #pragma unroll
    for (int j = 0; j < 4; ++j) {
      Ob[(size_t)(b * SEQ + qm) * D_MODEL + h * D_K + j * 16 + ln] =
          (unsigned short)f32_to_bf16_bits(accO[j][r] * inv);
    }
  }
}

// ---------------------------------------------------------------------------
extern "C" void kernel_launch(void* const* d_in, const int* in_sizes, int n_in,
                              void* d_out, int out_size, void* d_ws, size_t ws_size,
                              hipStream_t stream) {
  const float* q    = (const float*)d_in[0];
  const float* k    = (const float*)d_in[1];
  const float* v    = (const float*)d_in[2];
  const int*   mask = (const int*)  d_in[3];
  const float* Wq   = (const float*)d_in[4];
  const float* bq   = (const float*)d_in[5];
  const float* Wk   = (const float*)d_in[6];
  const float* bk   = (const float*)d_in[7];
  const float* Wv   = (const float*)d_in[8];
  const float* bv   = (const float*)d_in[9];
  const float* Wo   = (const float*)d_in[10];
  const float* bo   = (const float*)d_in[11];

  const size_t actE = (size_t)M_TOTAL * D_MODEL;      // 8M elements
  const size_t wE   = (size_t)D_MODEL * D_MODEL;      // 1M elements
  const size_t need = (7 * actE + 4 * wE) * sizeof(unsigned short);  // 120 MB
  if (ws_size < need) return;

  unsigned short* p   = (unsigned short*)d_ws;
  unsigned short* qb  = p;  p += actE;   // bf16 copies of inputs
  unsigned short* kb  = p;  p += actE;
  unsigned short* vb  = p;  p += actE;
  unsigned short* Wqb = p;  p += wE;     // bf16 copies of weights
  unsigned short* Wkb = p;  p += wE;
  unsigned short* Wvb = p;  p += wE;
  unsigned short* Wob = p;  p += wE;
  unsigned short* Qb  = p;  p += actE;   // projected Q/K (bf16, [b,s,1024])
  unsigned short* Kb  = p;  p += actE;
  unsigned short* Vt  = p;  p += actE;   // projected V (bf16, [b,h,dk,s])
  unsigned short* Ob  = p;  p += actE;   // attention output (bf16, [b,s,1024])

  // ---- one-time f32 -> bf16 conversion pre-pass ----
  const int actN4 = (int)(actE / 4), wN4 = (int)(wE / 4);
  const int ga = (actN4 + 255) / 256, gw = (wN4 + 255) / 256;
  convert_bf16_kernel<<<ga, 256, 0, stream>>>(q,  qb,  actN4);
  convert_bf16_kernel<<<ga, 256, 0, stream>>>(k,  kb,  actN4);
  convert_bf16_kernel<<<ga, 256, 0, stream>>>(v,  vb,  actN4);
  convert_bf16_kernel<<<gw, 256, 0, stream>>>(Wq, Wqb, wN4);
  convert_bf16_kernel<<<gw, 256, 0, stream>>>(Wk, Wkb, wN4);
  convert_bf16_kernel<<<gw, 256, 0, stream>>>(Wv, Wvb, wN4);
  convert_bf16_kernel<<<gw, 256, 0, stream>>>(Wo, Wob, wN4);

  // ---- projections ----
  dim3 gb(M_TOTAL / 128, D_MODEL / 64);   // (64, 16)
  gemm_bias_kernel<0><<<gb, 256, 0, stream>>>(qb, Wqb, bq, Qb);
  gemm_bias_kernel<0><<<gb, 256, 0, stream>>>(kb, Wkb, bk, Kb);
  gemm_bias_kernel<1><<<gb, 256, 0, stream>>>(vb, Wvb, bv, Vt);

  // ---- attention ----
  dim3 gf(SEQ / 128, BATCH * N_HEADS);    // (16, 64)
  flash_attn_kernel<<<gf, 256, 0, stream>>>(Qb, Kb, Vt, mask, Ob);

  // ---- output projection (f32 result) ----
  gemm_bias_kernel<2><<<gb, 256, 0, stream>>>(Ob, Wob, bo, d_out);
}
